// PointNetSVD_44667659878700
// MI455X (gfx1250) — compile-verified, matrix operand order checked by reading, not run on hardware
//
#include <hip/hip_runtime.h>
#include <hip/hip_bf16.h>

typedef __attribute__((ext_vector_type(16))) _Float16 v16h;
typedef __attribute__((ext_vector_type(8)))  _Float16 v8h;
typedef __attribute__((ext_vector_type(8)))  float    v8f;

union ABfrag { v16h v; v8h h[2]; };

// LDS layout (bytes):
//   w3f : 64nt * 4kc * 32lane * 16 f16  = 262144
//   w2f : 8nt  * 2kc * 32lane * 16 f16  =  16384
//   h1  : 16*64  f16                    =   2048
//   h2  : 16*128 f16                    =   4096
//   w1s : 192 f32, b1s: 64 f32, b2s: 128 f32, xt: 48 f32
#define LDS_BYTES (262144 + 16384 + 2048 + 4096 + 768 + 256 + 512 + 192)

extern "C" __global__ __launch_bounds__(256) void pointnet_fused(
    const float* __restrict__ x,  const float* __restrict__ W1, const float* __restrict__ b1,
    const float* __restrict__ W2, const float* __restrict__ b2,
    const float* __restrict__ W3, const float* __restrict__ b3,
    float* __restrict__ g)
{
    extern __shared__ char smem[];
    _Float16* w3f = (_Float16*)smem;                               // 131072 f16
    _Float16* w2f = (_Float16*)(smem + 262144);                    //   8192 f16
    _Float16* h1  = (_Float16*)(smem + 262144 + 16384);            //   1024 f16
    _Float16* h2  = (_Float16*)(smem + 262144 + 16384 + 2048);     //   2048 f16
    float*    w1s = (float*)   (smem + 262144 + 16384 + 2048 + 4096);
    float*    b1s = w1s + 192;
    float*    b2s = b1s + 64;
    float*    xt  = b2s + 128;   // 16 points x 3

    const int b    = blockIdx.x;
    const int tid  = threadIdx.x;
    const int wave = tid >> 5;
    const int lane = tid & 31;
    const int n16  = lane & 15;
    const int mhi  = (lane >> 4) & 1;     // which M-half (rows 0-7 vs 8-15) this lane accumulates

    // ---- one-time prepack: W3 (128x1024 f32 row-major) -> f16 B-fragments in LDS
    // fragment(nt,kc): 32 lanes x 16 f16, lane = n%16 + 16*((k%32)&8?1:0),
    // slot = (k&7) + 8*((k%32)>=16)  (matches 16-bit B-matrix VGPR layout, ISA 7.12.2)
    for (int idx = tid; idx < 128 * 1024; idx += 256) {
        int k = idx >> 10, n = idx & 1023;
        int nt = n >> 4, nn = n & 15;
        int kc = k >> 5, kk = k & 31;
        int fl = nn + (((kk >> 3) & 1) << 4);
        int slot = (kk & 7) + ((kk & 16) ? 8 : 0);
        w3f[(((nt << 2) + kc) << 9) + (fl << 4) + slot] = (_Float16)W3[idx];
    }
    // ---- prepack W2 (64x128) the same way
    for (int idx = tid; idx < 64 * 128; idx += 256) {
        int k = idx >> 7, n = idx & 127;
        int nt = n >> 4, nn = n & 15;
        int kc = k >> 5, kk = k & 31;
        int fl = nn + (((kk >> 3) & 1) << 4);
        int slot = (kk & 7) + ((kk & 16) ? 8 : 0);
        w2f[(((nt << 1) + kc) << 9) + (fl << 4) + slot] = (_Float16)W2[idx];
    }
    if (tid < 192) w1s[tid] = W1[tid];
    if (tid < 64)  b1s[tid] = b1[tid];
    if (tid < 128) b2s[tid] = b2[tid];
    __syncthreads();

    // wave handles layer2 N-tile = wave; layer3 N-tiles nt = wave + 8*j
    float bn2 = b2s[(wave << 4) + n16];
    float bn3[8], gacc[8];
#pragma unroll
    for (int j = 0; j < 8; ++j) {
        int nt = wave + (j << 3);
        bn3[j]  = b3[(nt << 4) + n16];
        gacc[j] = 0.0f;                      // ReLU outputs are >= 0, so 0 is the identity
    }

    for (int pt = 0; pt < 64; ++pt) {
        // (a) stage 16 points
        if (tid < 48) {
            int p = tid / 3, c = tid % 3;
            xt[tid] = x[((size_t)b * 1024 + (pt << 4) + p) * 3 + c];
        }
        __syncthreads();

        // (b) layer1 (3->64) VALU: 1024 outputs, 4 per thread -> h1 f16 [16][64]
#pragma unroll
        for (int r = 0; r < 4; ++r) {
            int o = (tid << 2) + r;
            int p = o >> 6, c = o & 63;
            float v = xt[p * 3 + 0] * w1s[c]
                    + xt[p * 3 + 1] * w1s[64 + c]
                    + xt[p * 3 + 2] * w1s[128 + c] + b1s[c];
            h1[o] = (_Float16)fmaxf(v, 0.0f);
        }
        __syncthreads();

        // (c) layer2 (64->128) WMMA: one 16x16 N-tile per wave, K=64 -> 2 WMMAs
        {
            v8f acc = {};
#pragma unroll
            for (int kc = 0; kc < 2; ++kc) {
                int koff = (kc << 5) + (mhi << 3);
                ABfrag a, bm;
                a.h[0] = *(const v8h*)&h1[(n16 << 6) + koff];
                a.h[1] = *(const v8h*)&h1[(n16 << 6) + koff + 16];
                bm.v   = *(const v16h*)&w2f[(((wave << 1) + kc) << 9) + (lane << 4)];
                acc = __builtin_amdgcn_wmma_f32_16x16x32_f16(
                          false, a.v, false, bm.v, (short)0, acc, false, false);
            }
#pragma unroll
            for (int r = 0; r < 8; ++r) {
                int m = r + (mhi << 3);
                h2[(m << 7) + (wave << 4) + n16] = (_Float16)fmaxf(acc[r] + bn2, 0.0f);
            }
        }
        __syncthreads();

        // (d) layer3 (128->1024) WMMA + fused bias/ReLU/max-pool
#pragma unroll
        for (int j = 0; j < 8; ++j) {
            int nt = wave + (j << 3);
            v8f acc = {};
#pragma unroll
            for (int kc = 0; kc < 4; ++kc) {
                int koff = (kc << 5) + (mhi << 3);
                ABfrag a, bm;
                a.h[0] = *(const v8h*)&h2[(n16 << 7) + koff];
                a.h[1] = *(const v8h*)&h2[(n16 << 7) + koff + 16];
                bm.v   = *(const v16h*)&w3f[(((nt << 2) + kc) << 9) + (lane << 4)];
                acc = __builtin_amdgcn_wmma_f32_16x16x32_f16(
                          false, a.v, false, bm.v, (short)0, acc, false, false);
            }
            float tmax = fmaxf(acc[0] + bn3[j], 0.0f);
#pragma unroll
            for (int r = 1; r < 8; ++r) tmax = fmaxf(tmax, fmaxf(acc[r] + bn3[j], 0.0f));
            // lanes l and l+16 hold the same N but M-halves 0-7 / 8-15: merge them
            tmax = fmaxf(tmax, __shfl_xor(tmax, 16, 32));
            gacc[j] = fmaxf(gacc[j], tmax);
        }
        __syncthreads();
    }

    if (lane < 16) {
#pragma unroll
        for (int j = 0; j < 8; ++j) {
            int nt = wave + (j << 3);
            g[(size_t)b * 1024 + (nt << 4) + n16] = gacc[j];
        }
    }
}

// ---- small head GEMMs (VALU; <0.5% of total FLOPs) ------------------------
extern "C" __global__ void dense_relu(const float* __restrict__ A, const float* __restrict__ W,
                                      const float* __restrict__ bias, float* __restrict__ out,
                                      int M, int K, int N)
{
    int o = blockIdx.x * blockDim.x + threadIdx.x;
    if (o >= M * N) return;
    int m = o / N, n = o % N;
    const float* a = A + (size_t)m * K;
    const float* w = W + n;
    float s = bias[n];
    for (int k = 0; k < K; k += 4) {
        s += a[k]     * w[(size_t)k * N];
        s += a[k + 1] * w[(size_t)(k + 1) * N];
        s += a[k + 2] * w[(size_t)(k + 2) * N];
        s += a[k + 3] * w[(size_t)(k + 3) * N];
    }
    out[o] = fmaxf(s, 0.0f);
}

// ---- final 256->9 + SO(3) projection, one thread per batch element --------
extern "C" __global__ void head_svd(const float* __restrict__ z2, const float* __restrict__ Wh3,
                                    const float* __restrict__ bh3, float* __restrict__ out)
{
    int bidx = blockIdx.x * blockDim.x + threadIdx.x;
    if (bidx >= 256) return;
    const float* zr = z2 + (size_t)bidx * 256;

    float M[9];
#pragma unroll
    for (int j = 0; j < 9; ++j) {
        float s = bh3[j];
        for (int k = 0; k < 256; ++k) s += zr[k] * Wh3[k * 9 + j];
        M[j] = s;
    }

    // S = M^T M (symmetric)
    float S[3][3];
#pragma unroll
    for (int i = 0; i < 3; ++i)
#pragma unroll
        for (int j = 0; j < 3; ++j) {
            float s = 0.0f;
            for (int k = 0; k < 3; ++k) s += M[k * 3 + i] * M[k * 3 + j];
            S[i][j] = s;
        }

    // cyclic Jacobi eigendecomposition of S; V accumulates rotations (det(V)=+1)
    float V[3][3] = {{1, 0, 0}, {0, 1, 0}, {0, 0, 1}};
    for (int sweep = 0; sweep < 8; ++sweep) {
        for (int pi = 0; pi < 3; ++pi) {
            int p = (pi == 2) ? 1 : 0;
            int q = (pi == 0) ? 1 : 2;
            float apq = S[p][q];
            if (fabsf(apq) > 1e-12f) {
                float theta = (S[q][q] - S[p][p]) / (2.0f * apq);
                float t = ((theta >= 0.0f) ? 1.0f : -1.0f)
                          / (fabsf(theta) + sqrtf(theta * theta + 1.0f));
                float c = 1.0f / sqrtf(t * t + 1.0f);
                float s = t * c;
                float app = S[p][p], aqq = S[q][q];
                S[p][p] = app - t * apq;
                S[q][q] = aqq + t * apq;
                S[p][q] = S[q][p] = 0.0f;
                int r = 3 - p - q;
                float arp = S[r][p], arq = S[r][q];
                S[r][p] = S[p][r] = c * arp - s * arq;
                S[r][q] = S[q][r] = s * arp + c * arq;
                for (int i = 0; i < 3; ++i) {
                    float vp = V[i][p], vq = V[i][q];
                    V[i][p] = c * vp - s * vq;
                    V[i][q] = s * vp + c * vq;
                }
            }
        }
    }

    // sort eigenvalues descending, track swap parity so det(V) stays +1
    int o0 = 0, o1 = 1, o2 = 2, par = 0, tmpi;
    if (S[o0][o0] < S[o1][o1]) { tmpi = o0; o0 = o1; o1 = tmpi; par ^= 1; }
    if (S[o1][o1] < S[o2][o2]) { tmpi = o1; o1 = o2; o2 = tmpi; par ^= 1; }
    if (S[o0][o0] < S[o1][o1]) { tmpi = o0; o0 = o1; o1 = tmpi; par ^= 1; }
    float v0[3], v1[3], v2[3];
    for (int i = 0; i < 3; ++i) { v0[i] = V[i][o0]; v1[i] = V[i][o1]; v2[i] = V[i][o2]; }
    if (par) { v2[0] = -v2[0]; v2[1] = -v2[1]; v2[2] = -v2[2]; }   // restore det(V)=+1

    // u0 = normalize(M v0); u1 = Gram-Schmidt(M v1); u2 = u0 x u1 (= det(U)*u_3)
    float u0[3], u1[3], w[3];
    for (int i = 0; i < 3; ++i)
        u0[i] = M[i * 3 + 0] * v0[0] + M[i * 3 + 1] * v0[1] + M[i * 3 + 2] * v0[2];
    float n0 = sqrtf(u0[0] * u0[0] + u0[1] * u0[1] + u0[2] * u0[2] + 1e-20f);
    for (int i = 0; i < 3; ++i) u0[i] /= n0;
    for (int i = 0; i < 3; ++i)
        u1[i] = M[i * 3 + 0] * v1[0] + M[i * 3 + 1] * v1[1] + M[i * 3 + 2] * v1[2];
    float d01 = u0[0] * u1[0] + u0[1] * u1[1] + u0[2] * u1[2];
    for (int i = 0; i < 3; ++i) u1[i] -= d01 * u0[i];
    float n1 = sqrtf(u1[0] * u1[0] + u1[1] * u1[1] + u1[2] * u1[2] + 1e-20f);
    for (int i = 0; i < 3; ++i) u1[i] /= n1;
    w[0] = u0[1] * u1[2] - u0[2] * u1[1];
    w[1] = u0[2] * u1[0] - u0[0] * u1[2];
    w[2] = u0[0] * u1[1] - u0[1] * u1[0];

    // R = u0 v0^T + u1 v1^T + (u0 x u1) v2^T = U diag(1,1,det(UV^T)) V^T
#pragma unroll
    for (int i = 0; i < 3; ++i)
#pragma unroll
        for (int j = 0; j < 3; ++j)
            out[bidx * 9 + i * 3 + j] = u0[i] * v0[j] + u1[i] * v1[j] + w[i] * v2[j];
}

extern "C" void kernel_launch(void* const* d_in, const int* in_sizes, int n_in,
                              void* d_out, int out_size, void* d_ws, size_t ws_size,
                              hipStream_t stream)
{
    const float* x   = (const float*)d_in[0];
    const float* W1  = (const float*)d_in[1];
    const float* b1  = (const float*)d_in[2];
    const float* W2  = (const float*)d_in[3];
    const float* b2  = (const float*)d_in[4];
    const float* W3  = (const float*)d_in[5];
    const float* b3  = (const float*)d_in[6];
    const float* Wh1 = (const float*)d_in[7];
    const float* bh1 = (const float*)d_in[8];
    const float* Wh2 = (const float*)d_in[9];
    const float* bh2 = (const float*)d_in[10];
    const float* Wh3 = (const float*)d_in[11];
    const float* bh3 = (const float*)d_in[12];

    float* g   = (float*)d_ws;            // [256][1024]
    float* z1  = g + 256 * 1024;          // [256][512]
    float* z2  = z1 + 256 * 512;          // [256][256]
    float* out = (float*)d_out;           // [256][3][3]

    pointnet_fused<<<256, 256, LDS_BYTES, stream>>>(x, W1, b1, W2, b2, W3, b3, g);
    dense_relu<<<(256 * 512) / 256, 256, 0, stream>>>(g, Wh1, bh1, z1, 256, 1024, 512);
    dense_relu<<<(256 * 256) / 256, 256, 0, stream>>>(z1, Wh2, bh2, z2, 256, 512, 256);
    head_svd<<<4, 64, 0, stream>>>(z2, Wh3, bh3, out);
}
